// MultiHeadAttention_58428735095200
// MI455X (gfx1250) — matrix operand order repaired and checked by reading, not compiled, not run on hardware
//
#include <hip/hip_runtime.h>

// ---------------- problem constants ----------------
constexpr int DM   = 1024;   // d_model (K and N of projection GEMMs)
constexpr int SEQ  = 2048;   // sequence length
constexpr int BSZ  = 2;      // batch
constexpr int MTOT = BSZ * SEQ; // 4096 rows for projection GEMMs
constexpr int NH   = 16;
constexpr int HD   = 64;

// ---------------- WMMA types ----------------
typedef __attribute__((ext_vector_type(16))) __bf16 v16bf;
typedef __attribute__((ext_vector_type(8)))  float  v8f;
typedef __attribute__((ext_vector_type(4)))  unsigned int u32x4;

union FragBF {
  u32x4 q[2];   // two 16B halves
  v16bf v;      // 16 bf16 per lane
};

__device__ __forceinline__ unsigned short f2bf(float f) {
  unsigned int u = __float_as_uint(f);
  u += 0x7FFFu + ((u >> 16) & 1u);   // round-to-nearest-even
  return (unsigned short)(u >> 16);
}

// ---------------- fp32 -> bf16 elementwise (x) ----------------
__global__ __launch_bounds__(256)
void cvt_bf16(const float* __restrict__ src, unsigned short* __restrict__ dst, int n) {
  int base = (blockIdx.x * blockDim.x + threadIdx.x) * 4;
  if (base + 3 < n) {
    float4 f = *(const float4*)&src[base];
    uint2 u;
    u.x = (unsigned)f2bf(f.x) | ((unsigned)f2bf(f.y) << 16);
    u.y = (unsigned)f2bf(f.z) | ((unsigned)f2bf(f.w) << 16);
    *(uint2*)(dst + base) = u;
  }
}

// ---------------- fp32 [K,N] -> bf16 [N,K] transpose (weights) ----------------
__global__ __launch_bounds__(1024)
void transpose_to_bf16(const float* __restrict__ W, unsigned short* __restrict__ Wt) {
  __shared__ float tile[32][33];
  int n0 = blockIdx.x * 32, k0 = blockIdx.y * 32;
  int tx = threadIdx.x, ty = threadIdx.y;
  tile[ty][tx] = W[(size_t)(k0 + ty) * DM + n0 + tx];
  __syncthreads();
  Wt[(size_t)(n0 + ty) * DM + k0 + tx] = f2bf(tile[tx][ty]);
}

// ---------------- bf16 WMMA GEMM: out = alpha*(A[M,K] @ Bt[N,K]^T + bias) ----------------
// MODE 0: bf16 out [M,N]   MODE 1: bf16 out V-transposed [B,H,HD,S]   MODE 2: f32 out [M,N]
template<int MODE>
__global__ __launch_bounds__(256)
void gemm_bf16_wmma(const unsigned short* __restrict__ A,
                    const unsigned short* __restrict__ Bt,
                    const float* __restrict__ bias,
                    void* __restrict__ out,
                    float alpha)
{
  constexpr int K = DM, BM = 128, BN = 64, BK = 32;
  constexpr int LDT = 40;                      // ushort pitch = 80B (bank-conflict pad)
  __shared__ unsigned short sA[BM * LDT];      // 10240 B
  __shared__ unsigned short sB[BN * LDT];      // 5120 B

  const int n0 = blockIdx.x * BN;
  const int m0 = blockIdx.y * BM;
  const int t = threadIdx.x;
  const int wave = t >> 5, lane = t & 31;
  const int lane16 = lane & 15, half = lane >> 4;

  v8f acc[4] = {};

  for (int k0 = 0; k0 < K; k0 += BK) {
    __syncthreads();
    {   // stage tiles: A 128x32 (2 chunks/thread), B 64x32 (1 chunk/thread), 16B each
      int r = t >> 2, c = t & 3;
      *(u32x4*)&sA[(r      ) * LDT + c * 8] = *(const u32x4*)&A[(size_t)(m0 + r     ) * K + k0 + c * 8];
      *(u32x4*)&sA[(r + 64 ) * LDT + c * 8] = *(const u32x4*)&A[(size_t)(m0 + r + 64) * K + k0 + c * 8];
      *(u32x4*)&sB[(r      ) * LDT + c * 8] = *(const u32x4*)&Bt[(size_t)(n0 + r    ) * K + k0 + c * 8];
    }
    __syncthreads();

    FragBF a;                                  // A frag: M=lane16, 16 contiguous K per half
    int arow = wave * 16 + lane16;
    int kb = half * 16;
    a.q[0] = *(const u32x4*)&sA[arow * LDT + kb];
    a.q[1] = *(const u32x4*)&sA[arow * LDT + kb + 8];
#pragma unroll
    for (int j = 0; j < 4; ++j) {              // B frag: N=lane16, 16 contiguous K per half
      FragBF b;
      int brow = j * 16 + lane16;
      int koff = half * 16;
      b.q[0] = *(const u32x4*)&sB[brow * LDT + koff];
      b.q[1] = *(const u32x4*)&sB[brow * LDT + koff + 8];
      acc[j] = __builtin_amdgcn_wmma_f32_16x16x32_bf16(false, a.v, false, b.v,
                                                       (short)0, acc[j], false, false);
    }
  }

#pragma unroll
  for (int j = 0; j < 4; ++j) {
    int n = n0 + j * 16 + lane16;
    float bn = bias[n];
#pragma unroll
    for (int r = 0; r < 8; ++r) {              // C frag: M = r + 8*half
      int m = m0 + wave * 16 + half * 8 + r;
      float val = (acc[j][r] + bn) * alpha;
      if (MODE == 2) {
        ((float*)out)[(size_t)m * DM + n] = val;
      } else if (MODE == 0) {
        ((unsigned short*)out)[(size_t)m * DM + n] = f2bf(val);
      } else {                                  // V transposed: [B,H,HD,S]
        int b = m >> 11, s = m & (SEQ - 1);
        int h = n >> 6,  d = n & (HD - 1);
        ((unsigned short*)out)[((size_t)(b * NH + h) * HD + d) * SEQ + s] = f2bf(val);
      }
    }
  }
}

// ---------------- causal flash attention, bf16 WMMA, fp32 online softmax ----------------
__global__ __launch_bounds__(256)
void flash_attn_wmma(const unsigned short* __restrict__ Qb,  // [B,S,DM] bf16, pre-scaled
                     const unsigned short* __restrict__ Kb,  // [B,S,DM] bf16
                     const unsigned short* __restrict__ Vt,  // [B,H,HD,S] bf16
                     unsigned short* __restrict__ O)         // [B,S,DM] bf16
{
  constexpr int LDP = 40;                      // padded P pitch
  __shared__ unsigned short sP[8 * 16 * LDP];  // wave-private 16x32 bf16 P tiles

  const int bh = blockIdx.x;                   // 0..31
  const int b = bh >> 4, h = bh & 15;
  const int wave = threadIdx.x >> 5, lane = threadIdx.x & 31;
  const int lane16 = lane & 15, half = lane >> 4;
  const int q0 = blockIdx.y * 128 + wave * 16;

  const unsigned short* qbase = Qb + (size_t)(b * SEQ) * DM + h * HD;
  const unsigned short* kbase = Kb + (size_t)(b * SEQ) * DM + h * HD;
  const unsigned short* vbase = Vt + (size_t)bh * HD * SEQ;
  unsigned short* pbase = &sP[wave * 16 * LDP];

  // Q A-fragments for the two hd chunks (kept in registers whole kernel)
  FragBF qa[2];
  {
    int srow = q0 + lane16;
    int kb = half * 16;
#pragma unroll
    for (int c = 0; c < 2; ++c) {
      qa[c].q[0] = *(const u32x4*)&qbase[(size_t)srow * DM + c * 32 + kb];
      qa[c].q[1] = *(const u32x4*)&qbase[(size_t)srow * DM + c * 32 + kb + 8];
    }
  }

  v8f o[4] = {};
  float mrun[8], lrun[8];
#pragma unroll
  for (int r = 0; r < 8; ++r) { mrun[r] = -1e30f; lrun[r] = 0.f; }

  const int kend = q0 + 16;                    // causal: keys <= q0+15
  for (int kblk = 0; kblk < kend; kblk += 32) {
    // ---- S = Q K^T for 32 keys (two 16-key sub-tiles, 4 WMMAs) ----
    v8f s0 = {}, s1 = {};
#pragma unroll
    for (int c = 0; c < 2; ++c) {
      FragBF bk0, bk1;
      int key0 = kblk + lane16;
      int koff = c * 32 + half * 16;
      bk0.q[0] = *(const u32x4*)&kbase[(size_t)key0 * DM + koff];
      bk0.q[1] = *(const u32x4*)&kbase[(size_t)key0 * DM + koff + 8];
      bk1.q[0] = *(const u32x4*)&kbase[(size_t)(key0 + 16) * DM + koff];
      bk1.q[1] = *(const u32x4*)&kbase[(size_t)(key0 + 16) * DM + koff + 8];
      s0 = __builtin_amdgcn_wmma_f32_16x16x32_bf16(false, qa[c].v, false, bk0.v, (short)0, s0, false, false);
      s1 = __builtin_amdgcn_wmma_f32_16x16x32_bf16(false, qa[c].v, false, bk1.v, (short)0, s1, false, false);
    }

    // ---- causal mask (uniform branch; only diagonal blocks) ----
    if (kblk + 31 > q0) {
      int key = kblk + lane16;
      int qrow = q0 + half * 8;
#pragma unroll
      for (int r = 0; r < 8; ++r) {
        if (key      > qrow + r) s0[r] = -1e30f;
        if (key + 16 > qrow + r) s1[r] = -1e30f;
      }
    }

    // ---- online softmax: row stats live in the 16-lane half that owns the rows ----
    float rescale[8];
#pragma unroll
    for (int r = 0; r < 8; ++r) {
      float m = fmaxf(s0[r], s1[r]);
#pragma unroll
      for (int d = 1; d < 16; d <<= 1) m = fmaxf(m, __shfl_xor(m, d, 32));
      float mn = fmaxf(mrun[r], m);
      float sc = __expf(mrun[r] - mn);
      mrun[r] = mn;
      rescale[r] = sc;
      float p0 = __expf(s0[r] - mn);
      float p1 = __expf(s1[r] - mn);
      s0[r] = p0; s1[r] = p1;
      float rsum = p0 + p1;
#pragma unroll
      for (int d = 1; d < 16; d <<= 1) rsum += __shfl_xor(rsum, d, 32);
      lrun[r] = lrun[r] * sc + rsum;
    }
#pragma unroll
    for (int j = 0; j < 4; ++j)
#pragma unroll
      for (int r = 0; r < 8; ++r) o[j][r] *= rescale[r];

    // ---- C-layout -> A-layout for P via wave-private LDS tile ----
#pragma unroll
    for (int r = 0; r < 8; ++r) {
      int row = half * 8 + r;
      pbase[row * LDP + lane16]      = f2bf(s0[r]);
      pbase[row * LDP + 16 + lane16] = f2bf(s1[r]);
    }
    FragBF pf;
    {
      int row = lane16, kb = half * 16;
      pf.q[0] = *(const u32x4*)&pbase[row * LDP + kb];
      pf.q[1] = *(const u32x4*)&pbase[row * LDP + kb + 8];
    }

    // ---- O += P V (4 WMMAs over hd chunks) ----
#pragma unroll
    for (int j = 0; j < 4; ++j) {
      FragBF vf;
      int d = j * 16 + lane16;
      int koff = half * 16;
      vf.q[0] = *(const u32x4*)&vbase[(size_t)d * SEQ + kblk + koff];
      vf.q[1] = *(const u32x4*)&vbase[(size_t)d * SEQ + kblk + koff + 8];
      o[j] = __builtin_amdgcn_wmma_f32_16x16x32_bf16(false, pf.v, false, vf.v, (short)0, o[j], false, false);
    }
  }

  // ---- finalize: O /= l, store bf16 ----
  float inv[8];
#pragma unroll
  for (int r = 0; r < 8; ++r) inv[r] = 1.0f / lrun[r];
  unsigned short* obase = O + (size_t)(b * SEQ + q0) * DM + h * HD;
#pragma unroll
  for (int j = 0; j < 4; ++j)
#pragma unroll
    for (int r = 0; r < 8; ++r) {
      int row = half * 8 + r;
      obase[(size_t)row * DM + j * 16 + lane16] = f2bf(o[j][r] * inv[r]);
    }
}

// ---------------- host-side launcher ----------------
extern "C" void kernel_launch(void* const* d_in, const int* in_sizes, int n_in,
                              void* d_out, int out_size, void* d_ws, size_t ws_size,
                              hipStream_t stream) {
  (void)in_sizes; (void)n_in; (void)out_size; (void)ws_size;
  const float* x  = (const float*)d_in[0];
  const float* Wq = (const float*)d_in[1];
  const float* bq = (const float*)d_in[2];
  const float* Wk = (const float*)d_in[3];
  const float* bk = (const float*)d_in[4];
  const float* Wv = (const float*)d_in[5];
  const float* bv = (const float*)d_in[6];
  const float* Wo = (const float*)d_in[7];
  const float* bo = (const float*)d_in[8];

  char* ws = (char*)d_ws;
  size_t off = 0;
  auto alloc = [&](size_t bytes) { void* p = ws + off; off += (bytes + 255) & ~(size_t)255; return p; };
  unsigned short* xb   = (unsigned short*)alloc((size_t)MTOT * DM * 2);  // x bf16
  unsigned short* wqt  = (unsigned short*)alloc((size_t)DM * DM * 2);    // Wq^T bf16
  unsigned short* wkt  = (unsigned short*)alloc((size_t)DM * DM * 2);
  unsigned short* wvt  = (unsigned short*)alloc((size_t)DM * DM * 2);
  unsigned short* wot  = (unsigned short*)alloc((size_t)DM * DM * 2);
  unsigned short* qb   = (unsigned short*)alloc((size_t)MTOT * DM * 2);  // Q  [B,S,DM]
  unsigned short* kb   = (unsigned short*)alloc((size_t)MTOT * DM * 2);  // K  [B,S,DM]
  unsigned short* vtb  = (unsigned short*)alloc((size_t)MTOT * DM * 2);  // V^T [B,H,HD,S]
  unsigned short* attb = (unsigned short*)alloc((size_t)MTOT * DM * 2);  // attn out bf16

  // 1) precision conversion / weight transpose
  cvt_bf16<<<(MTOT * DM) / (256 * 4), 256, 0, stream>>>(x, xb, MTOT * DM);
  dim3 tb(32, 32), tg(DM / 32, DM / 32);
  transpose_to_bf16<<<tg, tb, 0, stream>>>(Wq, wqt);
  transpose_to_bf16<<<tg, tb, 0, stream>>>(Wk, wkt);
  transpose_to_bf16<<<tg, tb, 0, stream>>>(Wv, wvt);
  transpose_to_bf16<<<tg, tb, 0, stream>>>(Wo, wot);

  // 2) QKV projections (Q pre-scaled by 1/sqrt(hd); V written transposed)
  dim3 gg(DM / 64, MTOT / 128), gb(256);
  gemm_bf16_wmma<0><<<gg, gb, 0, stream>>>(xb, wqt, bq, qb, 0.125f);
  gemm_bf16_wmma<0><<<gg, gb, 0, stream>>>(xb, wkt, bk, kb, 1.0f);
  gemm_bf16_wmma<1><<<gg, gb, 0, stream>>>(xb, wvt, bv, vtb, 1.0f);

  // 3) causal flash attention
  flash_attn_wmma<<<dim3(BSZ * NH, SEQ / 128), 256, 0, stream>>>(qb, kb, vtb, attb);

  // 4) output projection -> fp32
  gemm_bf16_wmma<2><<<gg, gb, 0, stream>>>(attb, wot, bo, d_out, 1.0f);
}
